// WindowedAttn_1425929142767
// MI455X (gfx1250) — matrix operand order repaired and checked by reading, not compiled, not run on hardware
//
#include <hip/hip_runtime.h>
#include <hip/hip_bf16.h>

// ---------------------------------------------------------------------------
// Problem constants (from reference): B=2, S=4096, D=2048, H=16, dh=128, W=512
// ---------------------------------------------------------------------------
#define D_MODEL   2048
#define N3        6144         // 3*D_MODEL
#define N_HEADS   16
#define D_HEAD    128
#define WINDOW    512
#define NWIN      8            // 4096 / 512
#define MROWS     8192         // B*S
#define ATT_SCALE 0.08838834764831845f  // 1/sqrt(128)

typedef __attribute__((ext_vector_type(16))) __bf16          v16bf;
typedef __attribute__((ext_vector_type(8)))  float           v8f;
typedef __attribute__((ext_vector_type(8)))  unsigned short  ush8;

union bfrag { v16bf v; ush8 h[2]; };

// ---------------------------------------------------------------------------
// CDNA5 async global->LDS copies (ASYNCcnt path) via inline asm.
// VFLAT async form: global_load_async_to_lds_bN  vdst(lds byte addr), vaddr, off
// Flat shared pointers carry the LDS byte offset in their low 32 bits
// ---------------------------------------------------------------------------
__device__ inline void async_ld128(unsigned short* lds, const unsigned short* g) {
    unsigned int  l = (unsigned int)(unsigned long long)lds;
    unsigned long long a = (unsigned long long)g;
    asm volatile("global_load_async_to_lds_b128 %0, %1, off"
                 :: "v"(l), "v"(a) : "memory");
}

__device__ inline void async_wait() {
#if __has_builtin(__builtin_amdgcn_s_wait_asynccnt)
    __builtin_amdgcn_s_wait_asynccnt(0);
#else
    asm volatile("s_wait_asynccnt 0x0" ::: "memory");
#endif
}

__device__ inline unsigned short f2bf(float f) {
    unsigned int u = __float_as_uint(f);
    unsigned int r = u + 0x7FFFu + ((u >> 16) & 1u);   // round-to-nearest-even
    return (unsigned short)(r >> 16);
}

__device__ inline v8f vzero8() {
    v8f z;
#pragma unroll
    for (int i = 0; i < 8; ++i) z[i] = 0.0f;
    return z;
}

__device__ inline v8f wmma_bf16(const bfrag& a, const bfrag& b, v8f c) {
    // (neg_a, A, neg_b, B, c_mod, C, reuse_a, reuse_b)
    return __builtin_amdgcn_wmma_f32_16x16x32_bf16(false, a.v, false, b.v,
                                                   (short)0, c, false, false);
}

// ---------------------------------------------------------------------------
// Kernel 1: f32 -> bf16 conversion (grid-stride)
// ---------------------------------------------------------------------------
__global__ void cvt_bf16_kernel(const float* __restrict__ in,
                                unsigned short* __restrict__ out, int n) {
    int i = blockIdx.x * blockDim.x + threadIdx.x;
    int stride = gridDim.x * blockDim.x;
    for (; i < n; i += stride) out[i] = f2bf(in[i]);
}

// ---------------------------------------------------------------------------
// Kernel 2: bf16 GEMM  C[M,N] = A[M,K] @ B[K,N] + bias   (f32 accumulate)
// 256 threads = 8 waves (2x4 wave grid); block tile 128x128, K step 32.
// Double-buffered LDS: async A-copies and B global loads for tile k+1 are
// issued before the WMMAs of tile k; one barrier per K step.
// ---------------------------------------------------------------------------
template <int OUT_BF16>
__global__ __launch_bounds__(256)
void gemm_bf16_kernel(const unsigned short* __restrict__ A,
                      const unsigned short* __restrict__ Bm,
                      const float* __restrict__ bias,
                      void* __restrict__ C,
                      int M, int N, int K) {
    __shared__ unsigned short Alds[2][128][48];   // [buf][m][k], 96B row stride
    __shared__ unsigned short Blds[2][128][48];   // [buf][n][k]  (B^T)

    const int tid  = threadIdx.x;
    const int lane = tid & 31;
    const int w    = tid >> 5;
    const int wm   = w >> 2;          // 0..1  -> 64 rows each
    const int wn   = w & 3;           // 0..3  -> 32 cols each
    const int lr   = lane & 15;
    const int kh   = lane >> 4;       // lane half selects K sub-chunk
    const int m0   = blockIdx.y * 128;
    const int n0   = blockIdx.x * 128;

    v8f acc[4][2];
#pragma unroll
    for (int mt = 0; mt < 4; ++mt)
#pragma unroll
        for (int nt = 0; nt < 2; ++nt) acc[mt][nt] = vzero8();

    const int arow = tid >> 1, ach = (tid & 1) * 16;       // A staging coords
    const int bkr  = tid >> 3, bcb = (tid & 7) * 16;       // B staging coords
    const unsigned short* arowp = A  + (size_t)(m0 + arow) * K + ach;
    const unsigned short* bkrp  = Bm + (size_t)bkr * N + n0 + bcb;

    // ---- prologue: stage K-tile 0 into buffer 0 ----
    async_ld128(&Alds[0][arow][ach],     arowp);
    async_ld128(&Alds[0][arow][ach + 8], arowp + 8);
    {
        ush8 b0 = *(const ush8*)bkrp;
        ush8 b1 = *(const ush8*)(bkrp + 8);
#pragma unroll
        for (int i = 0; i < 8; ++i) Blds[0][bcb + i][bkr]     = b0[i];
#pragma unroll
        for (int i = 0; i < 8; ++i) Blds[0][bcb + 8 + i][bkr] = b1[i];
    }
    async_wait();
    __syncthreads();

    const int KT = K >> 5;
    for (int kt = 0; kt < KT; ++kt) {
        const int cur = kt & 1, nxt = cur ^ 1;
        const int kk2 = (kt + 1) << 5;
        const bool have_next = (kt + 1 < KT);   // block-uniform

        // ---- issue next tile's loads before computing current tile ----
        ush8 b0, b1;
        if (have_next) {
            async_ld128(&Alds[nxt][arow][ach],     arowp + kk2);
            async_ld128(&Alds[nxt][arow][ach + 8], arowp + kk2 + 8);
            const unsigned short* bsrc = bkrp + (size_t)kk2 * N;
            b0 = *(const ush8*)bsrc;
            b1 = *(const ush8*)(bsrc + 8);
        }

        // ---- fragments per documented VGPR layouts, then 8 WMMAs ----
        bfrag afr[4], bfr[2];
#pragma unroll
        for (int mt = 0; mt < 4; ++mt) {
            int row = wm * 64 + mt * 16 + lr;              // A: lane%16 = M row
            afr[mt].h[0] = *(const ush8*)&Alds[cur][row][kh * 8];
            afr[mt].h[1] = *(const ush8*)&Alds[cur][row][16 + kh * 8];
        }
#pragma unroll
        for (int nt = 0; nt < 2; ++nt) {
            int col = wn * 32 + nt * 16 + lr;              // B: lane%16 = N col
            bfr[nt].h[0] = *(const ush8*)&Blds[cur][col][kh * 8];
            bfr[nt].h[1] = *(const ush8*)&Blds[cur][col][16 + kh * 8];
        }
#pragma unroll
        for (int mt = 0; mt < 4; ++mt)
#pragma unroll
            for (int nt = 0; nt < 2; ++nt)
                acc[mt][nt] = wmma_bf16(afr[mt], bfr[nt], acc[mt][nt]);

        // ---- finish staging next tile, single barrier per iteration ----
        if (have_next) {
#pragma unroll
            for (int i = 0; i < 8; ++i) Blds[nxt][bcb + i][bkr]     = b0[i];
#pragma unroll
            for (int i = 0; i < 8; ++i) Blds[nxt][bcb + 8 + i][bkr] = b1[i];
            async_wait();
        }
        __syncthreads();
    }

    // ---- epilogue: bias + store (compile-time output type) ----
#pragma unroll
    for (int nt = 0; nt < 2; ++nt) {
        int col = n0 + wn * 32 + nt * 16 + lr;
        float bv = bias[col];
#pragma unroll
        for (int mt = 0; mt < 4; ++mt) {
#pragma unroll
            for (int r = 0; r < 8; ++r) {
                int row = m0 + wm * 64 + mt * 16 + r + 8 * kh;  // C layout rows
                float v = acc[mt][nt][r] + bv;
                if (OUT_BF16) ((unsigned short*)C)[(size_t)row * N + col] = f2bf(v);
                else          ((float*)C)[(size_t)row * N + col] = v;
            }
        }
    }
}

// ---------------------------------------------------------------------------
// Kernel 3: windowed causal flash attention, one WG per (b, head, window).
// 1024 threads = 32 waves; wave w owns query rows [16w, 16w+16).
// Q for the whole window staged once in LDS; K/V chunks of 32 keys staged per
// iteration (K async, V transposed). Causal skip is a readfirstlane-uniform
// scalar branch so EXEC stays all-ones around the WMMAs.
// ---------------------------------------------------------------------------
__global__ __launch_bounds__(1024)
void wattn_kernel(const unsigned short* __restrict__ qkv,  // [8192, 6144] bf16
                  unsigned short* __restrict__ out) {      // [8192, 2048] bf16
    __shared__ unsigned short Qlds[512][136];         // [row][feat], 272B stride
    __shared__ unsigned short Klds[32][144];          // [key][feat], 288B stride
    __shared__ unsigned short Vt[128][48];            // V^T: [dh][key]
    __shared__ unsigned short Plds[32][16][32];       // per-wave P scratch

    const int tid  = threadIdx.x;
    const int lane = tid & 31;
    const int w    = tid >> 5;        // wave 0..31
    const int lr   = lane & 15;
    const int kh   = lane >> 4;
    const int wu   = __builtin_amdgcn_readfirstlane(w);  // SGPR wave id

    const int bid = blockIdx.x;                 // 0..255
    const int win = bid & 7;
    const int hh  = (bid >> 3) & 15;
    const int bb  = bid >> 7;
    const int rowbase = bb * 4096 + win * WINDOW;

    // ---- stage whole-window Q tile (512x128) into LDS via async copies ----
    {
        int qrow = tid >> 1;                    // 0..511
        int qch  = (tid & 1) * 64;              // column half
        const unsigned short* qsrc =
            qkv + (size_t)(rowbase + qrow) * N3 + hh * D_HEAD + qch;
#pragma unroll
        for (int i = 0; i < 8; ++i)
            async_ld128(&Qlds[qrow][qch + i * 8], qsrc + i * 8);
    }

    float mst[8], lst[8];
    v8f o[8];
#pragma unroll
    for (int r = 0; r < 8; ++r) { mst[r] = -1e30f; lst[r] = 0.0f; }
#pragma unroll
    for (int nt = 0; nt < 8; ++nt) o[nt] = vzero8();

    const int qr = w * 16 + lr;                 // this lane's Q row in window

    async_wait();
    __syncthreads();

    for (int j = 0; j < 16; ++j) {              // 16 key chunks of 32
        // ---- stage K (async b128, threads 0-511) / V^T (threads 512-1023) ----
        if (tid < 512) {
            int key = tid >> 4;                 // 0..31
            int cb  = (tid & 15) * 8;           // 0..120 step 8
            async_ld128(&Klds[key][cb],
                        qkv + (size_t)(rowbase + j * 32 + key) * N3 +
                            hh * D_HEAD + D_MODEL + cb);
        } else {
            int t2  = tid - 512;
            int key = t2 >> 4;
            int cb  = (t2 & 15) * 8;
            ush8 vv = *(const ush8*)(qkv + (size_t)(rowbase + j * 32 + key) * N3 +
                                     hh * D_HEAD + 2 * D_MODEL + cb);
#pragma unroll
            for (int i = 0; i < 8; ++i) Vt[cb + i][key] = vv[i];
        }
        async_wait();
        __syncthreads();

        if (j * 32 <= wu * 16 + 15) {           // scalar (SGPR) causal skip
            // ---- S = Q @ K^T for 16x32 score tile ----
            v8f s0 = vzero8(), s1 = vzero8();
#pragma unroll
            for (int kc = 0; kc < 4; ++kc) {
                bfrag qf, kf0, kf1;
                qf.h[0]  = *(const ush8*)&Qlds[qr][kc * 32 + kh * 8];
                qf.h[1]  = *(const ush8*)&Qlds[qr][kc * 32 + 16 + kh * 8];
                kf0.h[0] = *(const ush8*)&Klds[lr][kc * 32 + kh * 8];
                kf0.h[1] = *(const ush8*)&Klds[lr][kc * 32 + 16 + kh * 8];
                kf1.h[0] = *(const ush8*)&Klds[16 + lr][kc * 32 + kh * 8];
                kf1.h[1] = *(const ush8*)&Klds[16 + lr][kc * 32 + 16 + kh * 8];
                s0 = wmma_bf16(qf, kf0, s0);
                s1 = wmma_bf16(qf, kf1, s1);
            }

            // ---- online softmax (per C-layout row: VGPR r, lane half kh) ----
            v8f av;                             // per-row rescale factors
#pragma unroll
            for (int r = 0; r < 8; ++r) {
                int rloc = r + 8 * kh;          // row within this wave's tile
                int row  = w * 16 + rloc;       // row within window
                int key0 = j * 32 + lr;
                int key1 = key0 + 16;
                float v0 = s0[r] * ATT_SCALE;
                float v1 = s1[r] * ATT_SCALE;
                if (key0 > row) v0 = -1e30f;
                if (key1 > row) v1 = -1e30f;
                float mx = fmaxf(v0, v1);
                mx = fmaxf(mx, __shfl_xor(mx, 1, 16));
                mx = fmaxf(mx, __shfl_xor(mx, 2, 16));
                mx = fmaxf(mx, __shfl_xor(mx, 4, 16));
                mx = fmaxf(mx, __shfl_xor(mx, 8, 16));
                float mnew  = fmaxf(mst[r], mx);
                float alpha = __expf(mst[r] - mnew);
                float p0 = __expf(v0 - mnew);
                float p1 = __expf(v1 - mnew);
                float ps = p0 + p1;
                ps += __shfl_xor(ps, 1, 16);
                ps += __shfl_xor(ps, 2, 16);
                ps += __shfl_xor(ps, 4, 16);
                ps += __shfl_xor(ps, 8, 16);
                lst[r] = lst[r] * alpha + ps;
                mst[r] = mnew;
                av[r]  = alpha;
                Plds[w][rloc][lr]      = f2bf(p0);
                Plds[w][rloc][16 + lr] = f2bf(p1);
            }
#pragma unroll
            for (int nt = 0; nt < 8; ++nt) o[nt] *= av;   // vector rescale

            // ---- O += P @ V  (P re-read in A-matrix layout) ----
            bfrag pf;
            pf.h[0] = *(const ush8*)&Plds[w][lr][kh * 8];
            pf.h[1] = *(const ush8*)&Plds[w][lr][16 + kh * 8];
#pragma unroll
            for (int nt = 0; nt < 8; ++nt) {
                bfrag vf;
                int dc = nt * 16 + lr;
                vf.h[0] = *(const ush8*)&Vt[dc][kh * 8];
                vf.h[1] = *(const ush8*)&Vt[dc][16 + kh * 8];
                o[nt] = wmma_bf16(pf, vf, o[nt]);
            }
        }
        __syncthreads();
    }

    // ---- epilogue: O /= l, store bf16 to [8192, 2048] ----
#pragma unroll
    for (int r = 0; r < 8; ++r) {
        float linv = 1.0f / lst[r];
        int row = rowbase + w * 16 + r + 8 * kh;
        size_t ob = (size_t)row * D_MODEL + hh * D_HEAD + lr;
#pragma unroll
        for (int nt = 0; nt < 8; ++nt)
            out[ob + nt * 16] = f2bf(o[nt][r] * linv);
    }
}

// ---------------------------------------------------------------------------
// Host orchestration
// ---------------------------------------------------------------------------
extern "C" void kernel_launch(void* const* d_in, const int* in_sizes, int n_in,
                              void* d_out, int out_size, void* d_ws, size_t ws_size,
                              hipStream_t stream) {
    const float* x     = (const float*)d_in[0];   // [2,4096,2048]
    const float* W_qkv = (const float*)d_in[1];   // [2048,6144]
    const float* b_qkv = (const float*)d_in[2];   // [6144]
    const float* W_out = (const float*)d_in[3];   // [2048,2048]
    const float* b_out = (const float*)d_in[4];   // [2048]
    float* outp = (float*)d_out;                  // [2,4096,2048] f32

    char* ws = (char*)d_ws;
    unsigned short* x_bf    = (unsigned short*)(ws);                    // 33.5 MB
    unsigned short* wqkv_bf = (unsigned short*)(ws + 33554432);         // 25.2 MB
    unsigned short* wout_bf = (unsigned short*)(ws + 58720256);         //  8.4 MB
    unsigned short* qkv_bf  = (unsigned short*)(ws + 67108864);         // 100.7 MB
    unsigned short* attn_bf = (unsigned short*)(ws + 167772160);        // 33.5 MB

    // 1) downconvert inputs to bf16
    cvt_bf16_kernel<<<2048, 256, 0, stream>>>(x,     x_bf,    MROWS * D_MODEL);
    cvt_bf16_kernel<<<2048, 256, 0, stream>>>(W_qkv, wqkv_bf, D_MODEL * N3);
    cvt_bf16_kernel<<<1024, 256, 0, stream>>>(W_out, wout_bf, D_MODEL * D_MODEL);

    // 2) QKV projection: [8192,2048] @ [2048,6144] + b_qkv -> bf16
    gemm_bf16_kernel<1><<<dim3(N3 / 128, MROWS / 128), 256, 0, stream>>>(
        x_bf, wqkv_bf, b_qkv, qkv_bf, MROWS, N3, D_MODEL);

    // 3) windowed causal attention -> bf16 [8192,2048]
    wattn_kernel<<<2 * N_HEADS * NWIN, 1024, 0, stream>>>(qkv_bf, attn_bf);

    // 4) output projection: [8192,2048] @ [2048,2048] + b_out -> f32 d_out
    gemm_bf16_kernel<0><<<dim3(D_MODEL / 128, MROWS / 128), 256, 0, stream>>>(
        attn_bf, wout_bf, b_out, outp, MROWS, D_MODEL, D_MODEL);
}
